// ImprovedQuantumEncoder_32117765439979
// MI455X (gfx1250) — matrix operand order repaired and checked by reading, not compiled, not run on hardware
//
#include <hip/hip_runtime.h>

#define B_ROWS   131072
#define IN_DIM   192
#define NI       4
#define QDIM     64
#define HDIM     256      // NI*QDIM
#define OUT_DIM  256
#define TILE_M   16
#define WAVES    2        // waves (tiles) per block; 64-thread blocks

typedef __attribute__((ext_vector_type(16))) __bf16 v16bf;
typedef __attribute__((ext_vector_type(8)))  __bf16 v8bf;
typedef __attribute__((ext_vector_type(8)))  float  v8f;
typedef __attribute__((ext_vector_type(4)))  unsigned int u32x4;
typedef __attribute__((ext_vector_type(8)))  int i32x8;
typedef __attribute__((ext_vector_type(4)))  int i32x4;

#if __has_builtin(__builtin_amdgcn_tensor_load_to_lds) && __has_builtin(__builtin_amdgcn_s_wait_tensorcnt)
#define USE_TDM 1
#else
#define USE_TDM 0
#endif

union V16U { v16bf v; v8bf h[2]; };

// ws layout (ushort elements): W1[256*192] | Wqkv[192*64] | Wout[64*64] | Wproj[256*64] | Wpad[16*192]
#define WSU_W1    0
#define WSU_WQKV  (256*192)
#define WSU_WOUT  (WSU_WQKV + 192*64)
#define WSU_WPROJ (WSU_WOUT + 64*64)
#define WSU_WPAD  (WSU_WPROJ + 256*64)
#define WSU_END   (WSU_WPAD + 16*192)          // 84992 ushorts
#define WS_MW_BYTES 169984                      // 84992*2, 256B aligned

__device__ __forceinline__ unsigned short f2bf(float f) {
    unsigned int u = __float_as_uint(f);
    unsigned int r = (u + 0x7fffu + ((u >> 16) & 1u)) >> 16;   // RNE
    return (unsigned short)r;
}
__device__ __forceinline__ float bf2f(unsigned short u) {
    return __uint_as_float(((unsigned int)u) << 16);
}

// ---------------- prep: fp32 weights -> bf16 in ws, padded Wp tile, mixing softmax, zero scalars ----
__global__ void prep_kernel(const float* __restrict__ W1, const float* __restrict__ Wqkv,
                            const float* __restrict__ Wout, const float* __restrict__ Wproj,
                            const float* __restrict__ Wp, const float* __restrict__ mixing,
                            unsigned short* __restrict__ wbf, float* __restrict__ mw,
                            float* __restrict__ dscal) {
    const int n1 = 256*192, n2 = 192*64, n3 = 64*64, n4 = 256*64, n5 = 16*192;
    int idx = blockIdx.x * blockDim.x + threadIdx.x;
    int stride = gridDim.x * blockDim.x;
    for (int i = idx; i < n1 + n2 + n3 + n4 + n5; i += stride) {
        float v;
        if (i < n1)               v = W1[i];
        else if (i < n1+n2)       v = Wqkv[i - n1];
        else if (i < n1+n2+n3)    v = Wout[i - n1 - n2];
        else if (i < n1+n2+n3+n4) v = Wproj[i - n1 - n2 - n3];
        else {
            int k = i - (n1+n2+n3+n4);
            int rr = k / IN_DIM, cc = k % IN_DIM;
            v = (rr < NI) ? Wp[rr*IN_DIM + cc] : 0.f;          // zero-padded phase weights
        }
        wbf[i] = f2bf(v);
    }
    if (idx == 0) {
        for (int r = 0; r < 4; r++) {
            float mx = -1e30f;
            for (int c = 0; c < 4; c++) mx = fmaxf(mx, mixing[r*4+c]);
            float e[4], s = 0.f;
            for (int c = 0; c < 4; c++) { e[c] = expf(mixing[r*4+c] - mx); s += e[c]; }
            for (int c = 0; c < 4; c++) mw[r*4+c] = e[c] / s;
        }
        dscal[0] = 0.f;   // diversity accumulator
        dscal[1] = 0.f;   // phase variance accumulator
    }
}

// ---------------- fused encoder: one wave = one 16-row batch tile ----------------
__global__ __launch_bounds__(WAVES*32) void encoder_kernel(
    const float* __restrict__ x,
    const unsigned short* __restrict__ wbf,
    const float* __restrict__ b1, const float* __restrict__ ln_g, const float* __restrict__ ln_b,
    const float* __restrict__ bp,
    const float* __restrict__ bqkv, const float* __restrict__ bout_, const float* __restrict__ bproj,
    const float* __restrict__ mw,
    float* __restrict__ out, float* __restrict__ probs_o, float* __restrict__ coll_o,
    float* __restrict__ dscal)
{
    __shared__ __align__(32) unsigned short sh_a[WAVES][TILE_M*HDIM];    // bf16: x / mixed / attnv / collapsed
    __shared__ __align__(32) unsigned short sh_qkv[WAVES][TILE_M*768];   // bf16 qkv (4 interests * 192)
    __shared__ __align__(16) float          sh_h[WAVES][TILE_M*HDIM];    // f32: h / im / att
    __shared__ __align__(16) float          sh_xf[WAVES][TILE_M*IN_DIM]; // f32 x staging (TDM dest)
    __shared__                float          sh_s[WAVES][TILE_M*QDIM];   // phase / collapsed scratch

    const int w    = threadIdx.x >> 5;
    const int lane = threadIdx.x & 31;
    const int tile = blockIdx.x * WAVES + w;
    const int r0   = tile * TILE_M;

    unsigned short* A   = sh_a[w];
    unsigned short* QKV = sh_qkv[w];
    float*          H   = sh_h[w];
    float*          XF  = sh_xf[w];
    float*          S   = sh_s[w];

    const int row  = lane >> 1;      // per-row mapping (2 lanes per row)
    const int half = lane & 1;
    const int m16  = lane & 15;      // WMMA lane mapping
    const int hi16 = lane >> 4;

    // ---- stage 1: x tile -> LDS (TDM async tensor load) -> bf16 A operand ------
#if USE_TDM
    {
        // D# group0: count=1 | lds_addr | global_addr | type=2
        unsigned int ldsa = (unsigned int)(unsigned long long)(void*)XF;
        unsigned long long ga = (unsigned long long)(const void*)(x) + (unsigned long long)r0 * IN_DIM * 4ull;
        u32x4 g0;
        g0[0] = 1u;                                            // count=1, user mode
        g0[1] = ldsa;                                          // lds_addr (bytes)
        g0[2] = (unsigned int)(ga & 0xFFFFFFFFull);            // global_addr[31:0]
        g0[3] = (unsigned int)((ga >> 32) & 0x01FFFFFFull) | (2u << 30);  // global_addr[56:32] | type=2
        // D# group1: data_size=4B; tensor_dim0=192, tensor_dim1=16; tile 192x16; stride0=192
        i32x8 g1;
        g1[0] = 0x00020000;            // data_size=2 (4B), no mask/pad/iterate
        g1[1] = (int)(192u << 16);     // tensor_dim0[15:0] in [31:16]
        g1[2] = (int)(16u << 16);      // tensor_dim0 hi=0, tensor_dim1[15:0]=16
        g1[3] = (int)(192u << 16);     // tensor_dim1 hi=0, tile_dim0=192
        g1[4] = 16;                    // tile_dim1=16, tile_dim2=0
        g1[5] = 192;                   // tensor_dim0_stride lo
        g1[6] = 0;                     // stride0 hi, stride1 lo16
        g1[7] = 0;
        i32x4 gz4 = {0, 0, 0, 0};
        i32x8 gz8 = {0, 0, 0, 0, 0, 0, 0, 0};
        __builtin_amdgcn_tensor_load_to_lds(g0, g1, gz4, gz4, gz8, 0);
        __builtin_amdgcn_s_wait_tensorcnt((short)0);
    }
    __syncthreads();
    {
        const float4* xp = (const float4*)(XF + row*IN_DIM + half*96);
        unsigned short* ad = A + row*IN_DIM + half*96;
        for (int j = 0; j < 24; j++) {
            float4 t = xp[j];
            ad[4*j+0] = f2bf(t.x); ad[4*j+1] = f2bf(t.y);
            ad[4*j+2] = f2bf(t.z); ad[4*j+3] = f2bf(t.w);
        }
    }
#else
    {
        const float4* xp = (const float4*)(x + (size_t)(r0 + row)*IN_DIM + half*96);
        unsigned short* ad = A + row*IN_DIM + half*96;
        for (int j = 0; j < 24; j++) {
            float4 t = xp[j];
            ad[4*j+0] = f2bf(t.x); ad[4*j+1] = f2bf(t.y);
            ad[4*j+2] = f2bf(t.z); ad[4*j+3] = f2bf(t.w);
        }
    }
#endif
    __syncthreads();

    // ---- stage 2: GEMM1  h = x @ W1^T + b1, plus phase tile x @ Wpad^T ---------
    {
        const unsigned short* w1bf = wbf + WSU_W1;        // 256 x 192 row-major
        const int koff = hi16 ? 16 : 0;
        V16U a[6];
        #pragma unroll
        for (int kb = 0; kb < 6; kb++) {
            int base = m16*IN_DIM + kb*32 + koff;
            a[kb].h[0] = *(const v8bf*)(A + base);
            a[kb].h[1] = *(const v8bf*)(A + base + 8);
        }
        for (int nb = 0; nb < 16; nb++) {
            v8f c = {0.f,0.f,0.f,0.f,0.f,0.f,0.f,0.f};
            const int n = nb*16 + m16;
            #pragma unroll
            for (int kb = 0; kb < 6; kb++) {
                int k0 = kb*32 + (hi16 ? 16 : 0);
                v16bf b = *(const v16bf*)(w1bf + n*IN_DIM + k0);
                c = __builtin_amdgcn_wmma_f32_16x16x32_bf16(false, a[kb].v, false, b,
                                                            (short)0, c, false, false);
            }
            const int rbase = hi16 * 8;
            float bias = b1[n];
            #pragma unroll
            for (int vv = 0; vv < 8; vv++) H[(vv + rbase)*HDIM + n] = c[vv] + bias;
        }
        // phase tile: B = zero-padded Wp (16 x 192); only columns 0..3 meaningful
        {
            const unsigned short* wpad = wbf + WSU_WPAD;
            v8f c = {0.f,0.f,0.f,0.f,0.f,0.f,0.f,0.f};
            #pragma unroll
            for (int kb = 0; kb < 6; kb++) {
                int k0 = kb*32 + (hi16 ? 16 : 0);
                v16bf b = *(const v16bf*)(wpad + m16*IN_DIM + k0);
                c = __builtin_amdgcn_wmma_f32_16x16x32_bf16(false, a[kb].v, false, b,
                                                            (short)0, c, false, false);
            }
            const int rbase = hi16 * 8;
            if (m16 < NI) {
                float bias = bp[m16];
                #pragma unroll
                for (int vv = 0; vv < 8; vv++) S[(vv + rbase)*NI + m16] = c[vv] + bias;
            }
        }
    }
    __syncthreads();

    // ---- stage 3: phases, LayerNorm + GELU + phase gate + l2norm + mixing ------
    float pwv[4];
    {
        float phs[4], pm = 0.f;
        #pragma unroll
        for (int i = 0; i < 4; i++) {
            phs[i] = tanhf(S[row*NI + i]);
            pwv[i] = 1.0f / (1.0f + expf(-3.14159265358979f * phs[i]));
            pm += phs[i];
        }
        pm *= 0.25f;
        float pv = 0.f;
        #pragma unroll
        for (int i = 0; i < 4; i++) { float d = phs[i] - pm; pv += d*d; }
        pv *= (1.0f/3.0f);                         // ddof=1
        if (half == 0) atomicAdd(&dscal[1], pv * (1.0f / (float)B_ROWS));
    }
    {
        float* hrow = H + row*HDIM;
        const int d0 = half * 128;
        float s = 0.f, ss = 0.f;
        for (int d = 0; d < 128; d++) { float v = hrow[d0+d]; s += v; ss += v*v; }
        s  += __shfl_xor(s, 1, 32);
        ss += __shfl_xor(ss, 1, 32);
        float mu   = s * (1.0f/256.0f);
        float var  = ss * (1.0f/256.0f) - mu*mu;
        float rstd = rsqrtf(var + 1e-5f);
        float ssg[2] = {0.f, 0.f};
        for (int d = 0; d < 128; d++) {
            int gd = d0 + d;
            float v  = (hrow[gd] - mu) * rstd * ln_g[gd] + ln_b[gd];
            float ge = 0.5f * v * (1.0f + erff(v * 0.70710678118f));   // exact GELU
            float im = ge * pwv[gd >> 6];
            hrow[gd] = im;                       // unnormalized im back to LDS
            ssg[d >> 6] += im * im;
        }
        float sc0 = 1.0f / fmaxf(sqrtf(ssg[0]), 1e-12f);
        float sc1 = 1.0f / fmaxf(sqrtf(ssg[1]), 1e-12f);
        float os0 = __shfl_xor(sc0, 1, 32);
        float os1 = __shfl_xor(sc1, 1, 32);
        float scale[4];
        scale[2*half]         = sc0;  scale[2*half + 1]         = sc1;
        scale[2*(1-half)]     = os0;  scale[2*(1-half) + 1]     = os1;
        __syncthreads();             // all im values visible
        float mwr[2][4];
        #pragma unroll
        for (int t = 0; t < 2; t++) {
            int i = 2*half + t;
            #pragma unroll
            for (int j = 0; j < 4; j++) mwr[t][j] = mw[i*4 + j];
        }
        unsigned short* arow = A + row*HDIM;
        for (int d = 0; d < 64; d++) {
            float g0 = hrow[0*64+d]*scale[0], g1 = hrow[1*64+d]*scale[1];
            float g2 = hrow[2*64+d]*scale[2], g3 = hrow[3*64+d]*scale[3];
            #pragma unroll
            for (int t = 0; t < 2; t++) {
                int i = 2*half + t;
                float mv = mwr[t][0]*g0 + mwr[t][1]*g1 + mwr[t][2]*g2 + mwr[t][3]*g3;
                arow[i*64 + d] = f2bf(mv);      // mixed, bf16, stride 256
            }
        }
    }
    __syncthreads();

    // ---- stage 4: GEMM2  qkv = mixed @ Wqkv^T + bqkv (per interest) ------------
    {
        const unsigned short* wqkvbf = wbf + WSU_WQKV;       // 192 x 64
        const int koff = hi16 ? 16 : 0;
        const int bk   = hi16 ? 16 : 0;
        for (int i = 0; i < NI; i++) {
            V16U a2[2];
            #pragma unroll
            for (int kb = 0; kb < 2; kb++) {
                int base = m16*HDIM + i*64 + kb*32 + koff;
                a2[kb].h[0] = *(const v8bf*)(A + base);
                a2[kb].h[1] = *(const v8bf*)(A + base + 8);
            }
            for (int nb = 0; nb < 12; nb++) {
                v8f c = {0.f,0.f,0.f,0.f,0.f,0.f,0.f,0.f};
                const int n = nb*16 + m16;
                #pragma unroll
                for (int kb = 0; kb < 2; kb++) {
                    v16bf b = *(const v16bf*)(wqkvbf + n*64 + kb*32 + bk);
                    c = __builtin_amdgcn_wmma_f32_16x16x32_bf16(false, a2[kb].v, false, b,
                                                                (short)0, c, false, false);
                }
                const int rbase = hi16 * 8;
                float bias = bqkv[n];
                #pragma unroll
                for (int vv = 0; vv < 8; vv++)
                    QKV[(vv + rbase)*768 + i*192 + n] = f2bf(c[vv] + bias);
            }
        }
    }
    __syncthreads();

    // ---- stage 5: tiny 4x4 attention per row -----------------------------------
    {
        const unsigned short* qk = QKV + row*768;
        #pragma unroll
        for (int t = 0; t < 2; t++) {
            int i = 2*half + t;
            float a_[4];
            {
                float sc[4];
                #pragma unroll
                for (int j = 0; j < 4; j++) {
                    float dot = 0.f;
                    for (int d = 0; d < 64; d++)
                        dot += bf2f(qk[i*192 + d]) * bf2f(qk[j*192 + 64 + d]);
                    sc[j] = dot * 0.125f;                     // QDIM^-0.5
                }
                float mx = fmaxf(fmaxf(sc[0], sc[1]), fmaxf(sc[2], sc[3]));
                float sm = 0.f;
                #pragma unroll
                for (int j = 0; j < 4; j++) { a_[j] = expf(sc[j] - mx); sm += a_[j]; }
                float inv = 1.0f / sm;
                #pragma unroll
                for (int j = 0; j < 4; j++) a_[j] *= inv;
            }
            unsigned short* arow = A + row*HDIM + i*64;
            for (int d = 0; d < 64; d++) {
                float o = a_[0]*bf2f(qk[0*192+128+d]) + a_[1]*bf2f(qk[1*192+128+d])
                        + a_[2]*bf2f(qk[2*192+128+d]) + a_[3]*bf2f(qk[3*192+128+d]);
                arow[d] = f2bf(o);                            // attnv, bf16
            }
        }
    }
    __syncthreads();

    // ---- stage 6: GEMM3  att = attnv @ Wout^T + bout ---------------------------
    {
        const unsigned short* woutbf = wbf + WSU_WOUT;   // 64 x 64
        const int koff = hi16 ? 16 : 0;
        const int bk   = hi16 ? 16 : 0;
        for (int i = 0; i < NI; i++) {
            V16U a3[2];
            #pragma unroll
            for (int kb = 0; kb < 2; kb++) {
                int base = m16*HDIM + i*64 + kb*32 + koff;
                a3[kb].h[0] = *(const v8bf*)(A + base);
                a3[kb].h[1] = *(const v8bf*)(A + base + 8);
            }
            for (int nb = 0; nb < 4; nb++) {
                v8f c = {0.f,0.f,0.f,0.f,0.f,0.f,0.f,0.f};
                const int n = nb*16 + m16;
                #pragma unroll
                for (int kb = 0; kb < 2; kb++) {
                    v16bf b = *(const v16bf*)(woutbf + n*64 + kb*32 + bk);
                    c = __builtin_amdgcn_wmma_f32_16x16x32_bf16(false, a3[kb].v, false, b,
                                                                (short)0, c, false, false);
                }
                const int rbase = hi16 * 8;
                float bias = bout_[n];
                #pragma unroll
                for (int vv = 0; vv < 8; vv++)
                    H[(vv + rbase)*HDIM + i*64 + n] = c[vv] + bias;   // att, f32
            }
        }
    }
    __syncthreads();

    // ---- stage 7: norms -> probs -> collapsed -> diversity ---------------------
    {
        float* att = H + row*HDIM;
        const int i0 = 2*half, i1 = 2*half + 1;
        float ss0 = 0.f, ss1 = 0.f;
        for (int d = 0; d < 64; d++) {
            float v0 = att[i0*64 + d], v1 = att[i1*64 + d];
            ss0 += v0*v0; ss1 += v1*v1;
        }
        float os0 = __shfl_xor(ss0, 1, 32);
        float os1 = __shfl_xor(ss1, 1, 32);
        float nn[4];
        nn[i0] = sqrtf(ss0); nn[i1] = sqrtf(ss1);
        nn[2*(1-half)] = sqrtf(os0); nn[2*(1-half)+1] = sqrtf(os1);

        float p[4];
        {
            float mx = fmaxf(fmaxf(nn[0], nn[1]), fmaxf(nn[2], nn[3]));
            float sm = 0.f;
            #pragma unroll
            for (int j = 0; j < 4; j++) { p[j] = expf(nn[j] - mx); sm += p[j]; }
            float inv = 1.0f / sm;
            #pragma unroll
            for (int j = 0; j < 4; j++) p[j] *= inv;
            if (half == 0) {
                #pragma unroll
                for (int j = 0; j < 4; j++) probs_o[(size_t)(r0 + row)*4 + j] = p[j];
            }
        }
        if (half == 1)
            for (int d = 0; d < 64; d++)
                S[row*64 + d] = p[i0]*att[i0*64+d] + p[i1]*att[i1*64+d];
        __syncthreads();
        if (half == 0)
            for (int d = 0; d < 64; d++) {
                float cv = p[i0]*att[i0*64+d] + p[i1]*att[i1*64+d] + S[row*64 + d];
                coll_o[(size_t)(r0 + row)*64 + d] = cv;
                A[row*HDIM + d] = f2bf(cv);                  // collapsed, bf16, for GEMM4
            }
        // diversity: off-diagonal squared cosine sims
        float acc = 0.f;
        #pragma unroll
        for (int t = 0; t < 2; t++) {
            int i = 2*half + t;
            for (int j = 0; j < 4; j++) {
                if (j == i) continue;
                float dot = 0.f;
                for (int d = 0; d < 64; d++) dot += att[i*64+d] * att[j*64+d];
                float r = dot / (fmaxf(nn[i], 1e-12f) * fmaxf(nn[j], 1e-12f));
                acc += r*r;
            }
        }
        acc += __shfl_xor(acc, 1, 32);
        if (half == 0)
            atomicAdd(&dscal[0], acc * (1.0f / ((12.0f + 1e-8f) * (float)B_ROWS)));
        __syncthreads();
    }

    // ---- stage 8: GEMM4  output = collapsed @ Wproj^T + bproj -> global --------
    {
        const unsigned short* wprojbf = wbf + WSU_WPROJ;  // 256 x 64
        const int koff = hi16 ? 16 : 0;
        const int bk   = hi16 ? 16 : 0;
        V16U a4[2];
        #pragma unroll
        for (int kb = 0; kb < 2; kb++) {
            int base = m16*HDIM + kb*32 + koff;
            a4[kb].h[0] = *(const v8bf*)(A + base);
            a4[kb].h[1] = *(const v8bf*)(A + base + 8);
        }
        for (int nb = 0; nb < 16; nb++) {
            v8f c = {0.f,0.f,0.f,0.f,0.f,0.f,0.f,0.f};
            const int n = nb*16 + m16;
            #pragma unroll
            for (int kb = 0; kb < 2; kb++) {
                v16bf b = *(const v16bf*)(wprojbf + n*64 + kb*32 + bk);
                c = __builtin_amdgcn_wmma_f32_16x16x32_bf16(false, a4[kb].v, false, b,
                                                            (short)0, c, false, false);
            }
            const int rbase = hi16 * 8;
            float bias = bproj[n];
            #pragma unroll
            for (int vv = 0; vv < 8; vv++)
                out[(size_t)(r0 + vv + rbase)*OUT_DIM + n] = c[vv] + bias;
        }
    }
}

extern "C" void kernel_launch(void* const* d_in, const int* in_sizes, int n_in,
                              void* d_out, int out_size, void* d_ws, size_t ws_size,
                              hipStream_t stream) {
    const float* x      = (const float*)d_in[0];
    const float* W1     = (const float*)d_in[1];
    const float* b1     = (const float*)d_in[2];
    const float* ln_g   = (const float*)d_in[3];
    const float* ln_b   = (const float*)d_in[4];
    const float* Wp     = (const float*)d_in[5];
    const float* bp     = (const float*)d_in[6];
    const float* mixing = (const float*)d_in[7];
    const float* Wqkv   = (const float*)d_in[8];
    const float* bqkv   = (const float*)d_in[9];
    const float* Wout   = (const float*)d_in[10];
    const float* bout   = (const float*)d_in[11];
    const float* Wproj  = (const float*)d_in[12];
    const float* bproj  = (const float*)d_in[13];

    float* out     = (float*)d_out;
    float* probs_o = out + (size_t)B_ROWS * 256;
    float* coll_o  = probs_o + (size_t)B_ROWS * 4;
    float* dscal   = coll_o + (size_t)B_ROWS * 64;   // [diversity, phase_variance]

    unsigned short* wbf = (unsigned short*)d_ws;     // bf16 weights + padded Wp
    float* mw = (float*)((char*)d_ws + WS_MW_BYTES); // 4x4 softmaxed mixing

    prep_kernel<<<320, 256, 0, stream>>>(W1, Wqkv, Wout, Wproj, Wp, mixing, wbf, mw, dscal);

    encoder_kernel<<<B_ROWS / TILE_M / WAVES, WAVES * 32, 0, stream>>>(
        x, wbf, b1, ln_g, ln_b, bp, bqkv, bout, bproj, mw,
        out, probs_o, coll_o, dscal);
}
